// MultiHeadEFRALayer_14998025797907
// MI455X (gfx1250) — compile-verified, matrix operand order checked by reading, not run on hardware
//
#include <hip/hip_runtime.h>
#include <hip/hip_bf16.h>

// ---------------------------------------------------------------------------
// MultiHeadEFRA layer for MI455X (gfx1250, wave32, WMMA bf16,
// async direct-to-LDS double-buffered GEMM)
// D=256 H=4 DK=32 DV=64 DFF=1024 NTF=600 NG=8000
// ---------------------------------------------------------------------------

typedef __attribute__((ext_vector_type(16))) __bf16 v16bf;
typedef __attribute__((ext_vector_type(8)))  float  v8f;
typedef __attribute__((ext_vector_type(4)))  int    v4i_t;

#define C_D    256
#define C_H    4
#define C_DK   32
#define C_DFF  1024
#define C_NTF  600
#define C_NG   8000

#if __has_builtin(__builtin_amdgcn_global_load_async_to_lds_b128) && \
    __has_builtin(__builtin_amdgcn_s_wait_asynccnt)
#define HAVE_ASYNC_LDS 1
#else
#define HAVE_ASYNC_LDS 0
#endif

// ---------------- block reductions (256 threads) ----------------
__device__ __forceinline__ float block_sum256(float v, float* red, int tid) {
    red[tid] = v; __syncthreads();
    #pragma unroll
    for (int s = 128; s > 0; s >>= 1) {
        if (tid < s) red[tid] += red[tid + s];
        __syncthreads();
    }
    float r = red[0]; __syncthreads();
    return r;
}

__device__ __forceinline__ float block_max256(float v, float* red, int tid) {
    red[tid] = v; __syncthreads();
    #pragma unroll
    for (int s = 128; s > 0; s >>= 1) {
        if (tid < s) red[tid] = fmaxf(red[tid], red[tid + s]);
        __syncthreads();
    }
    float r = red[0]; __syncthreads();
    return r;
}

__device__ __forceinline__ float gelu_exact(float x) {
    return 0.5f * x * (1.0f + erff(x * 0.7071067811865476f));
}

// ---------------- fp32 -> bf16 conversion (4 elts / thread) ----------------
__global__ __launch_bounds__(256) void cvt_bf16_kernel(const float* __restrict__ x,
                                                       __bf16* __restrict__ y,
                                                       long n4) {
    long i = ((long)blockIdx.x * 256 + threadIdx.x) * 4;
    if (i < n4 * 4) {
        float4 v = *(const float4*)(x + i);
        y[i + 0] = (__bf16)v.x;
        y[i + 1] = (__bf16)v.y;
        y[i + 2] = (__bf16)v.z;
        y[i + 3] = (__bf16)v.w;
    }
}

// ---------------- LayerNorm: one block per row, D == 256 ----------------
__global__ __launch_bounds__(256) void ln_kernel(const float* __restrict__ x,
                                                 const float* __restrict__ g,
                                                 const float* __restrict__ b,
                                                 float* __restrict__ y,
                                                 __bf16* __restrict__ ybf) {
    const int row = blockIdx.x, tid = threadIdx.x;
    __shared__ float red[256];
    float v = x[(size_t)row * C_D + tid];
    float m = block_sum256(v, red, tid) * (1.0f / C_D);
    float d = v - m;
    float var = block_sum256(d * d, red, tid) * (1.0f / C_D);
    float r = d * rsqrtf(var + 1e-5f) * g[tid] + b[tid];
    if (y)   y[(size_t)row * C_D + tid] = r;
    if (ybf) ybf[(size_t)row * C_D + tid] = (__bf16)r;
}

// ---------------- gather z[tf_idx] for 3 channels (bf16 out) ----------------
__global__ __launch_bounds__(256) void gather3_kernel(const float* __restrict__ zs,
                                                      const float* __restrict__ ze,
                                                      const float* __restrict__ zt,
                                                      const int* __restrict__ idx,
                                                      __bf16* __restrict__ os,
                                                      __bf16* __restrict__ oe,
                                                      __bf16* __restrict__ ot) {
    const int t = blockIdx.x, tid = threadIdx.x;
    const size_t j = (size_t)idx[t] * C_D;
    os[(size_t)t * C_D + tid] = (__bf16)zs[j + tid];
    oe[(size_t)t * C_D + tid] = (__bf16)ze[j + tid];
    ot[(size_t)t * C_D + tid] = (__bf16)zt[j + tid];
}

// ---------------- bf16 WMMA GEMM (double-buffered, async-to-LDS) ----------
// C[M,N] = act( A x B^T + bias ) + resid;  A,B bf16 in memory, fp32 accumulate.
// TR==0: P(r,k) = P[r*ld + k] (k contiguous). CONTRACT: K % 32 == 0 at TR0
//        call sites (true here: K in {256,1024,8000}) -> no k-guard, pure
//        async b128 direct-to-LDS.
// TR==1: P(r,k) = P[k*ld + r] (r contiguous); guarded (serves K=600).
// All M,N,K are multiples of 8 -> vector-granular bounds. Row-clamped loads
// are safe because C rows >= M are never stored.

template <int TR>
__device__ __forceinline__ void load_tile(const __bf16* __restrict__ P, int ld,
                                          int base, int Rlim, int kt, int K,
                                          int tid, __bf16 (*S)[40]) {
    if (TR == 0) {
        const int lr = tid >> 2;            // 0..63 : tile row (m or n)
        const int lk = (tid & 3) * 8;       // 0,8,16,24
        int gr = base + lr;
        if (gr > Rlim - 1) gr = Rlim - 1;   // clamp: OOB C rows never stored
        const __bf16* src = P + (size_t)gr * ld + (kt + lk);
#if HAVE_ASYNC_LDS
        __attribute__((address_space(1))) v4i_t* gp =
            (__attribute__((address_space(1))) v4i_t*)(uintptr_t)src;
        __attribute__((address_space(3))) v4i_t* lp =
            (__attribute__((address_space(3))) v4i_t*)&S[lr][lk];
        __builtin_amdgcn_global_load_async_to_lds_b128(gp, lp, 0, 0);
#else
        *(uint4*)&S[lr][lk] = *(const uint4*)src;
#endif
        if (kt + lk + 64 < K) __builtin_prefetch(src + 64, 0, 1);
    } else {
        const int kk = tid >> 3;            // 0..31 : k within tile
        const int r8 = (tid & 7) * 8;       // 0,8,..,56 : row group
        const int gk = kt + kk;
        const int gr = base + r8;
        union { uint4 q; __bf16 h[8]; } v;
        v.q = (uint4){0u, 0u, 0u, 0u};
        if (gk < K && gr < Rlim) {
            const __bf16* src = P + (size_t)gk * ld + gr;
            v.q = *(const uint4*)src;
            if (gk + 32 < K) __builtin_prefetch(src + (size_t)32 * ld, 0, 1);
        }
        #pragma unroll
        for (int i = 0; i < 8; ++i) S[r8 + i][kk] = v.h[i];
    }
}

__device__ __forceinline__ v16bf frag_from_row(const __bf16* row, int khalf) {
    union { uint4 q; __bf16 h[8]; } lo, hi;
    lo.q = *(const uint4*)(row + khalf);        // K = khalf .. khalf+7
    hi.q = *(const uint4*)(row + 16 + khalf);   // K = 16+khalf .. +7
    v16bf f;
    #pragma unroll
    for (int i = 0; i < 8; ++i) { f[i] = lo.h[i]; f[8 + i] = hi.h[i]; }
    return f;
}

template <int AT, int BT>
__global__ __launch_bounds__(256) void gemm_bf16_kernel(
    const __bf16* __restrict__ A, int lda,
    const __bf16* __restrict__ B, int ldb,
    float* __restrict__ C, __bf16* __restrict__ Cbf, int ldc,
    const float* __restrict__ bias,
    const float* __restrict__ resid, int ldres,
    int M, int N, int K, int act) {

    __shared__ __align__(16) __bf16 As[2][64][40];
    __shared__ __align__(16) __bf16 Bs[2][64][40];

    const int tid  = threadIdx.x;
    const int lane = tid & 31;
    const int wave = tid >> 5;
    const int wm   = wave >> 1;           // 0..3  M sub-tile
    const int wn   = wave & 1;            // 0..1  N sub-tile (32 wide)
    const int m0   = blockIdx.y * 64;
    const int n0   = blockIdx.x * 64;
    const int lm   = lane & 15;
    const int khalf = (lane >> 4) * 8;

    v8f c0 = {};
    v8f c1 = {};

    // prologue: stage first K-tile into buffer 0
    load_tile<AT>(A, lda, m0, M, 0, K, tid, As[0]);
    load_tile<BT>(B, ldb, n0, N, 0, K, tid, Bs[0]);

    int cur = 0;
    for (int kt = 0; kt < K; kt += 32, cur ^= 1) {
#if HAVE_ASYNC_LDS
        __builtin_amdgcn_s_wait_asynccnt(0);   // buf[cur] DMA complete (this wave)
#endif
        __syncthreads();                       // all waves: buf[cur] ready,
                                               // buf[cur^1] fully consumed
        if (kt + 32 < K) {                     // overlap next tile with compute
            load_tile<AT>(A, lda, m0, M, kt + 32, K, tid, As[cur ^ 1]);
            load_tile<BT>(B, ldb, n0, N, kt + 32, K, tid, Bs[cur ^ 1]);
        }

        v16bf a  = frag_from_row(&As[cur][wm * 16 + lm][0], khalf);
        v16bf b0 = frag_from_row(&Bs[cur][wn * 32 + lm][0], khalf);
        v16bf b1 = frag_from_row(&Bs[cur][wn * 32 + 16 + lm][0], khalf);

        c0 = __builtin_amdgcn_wmma_f32_16x16x32_bf16(false, a, false, b0,
                                                     (short)0, c0, false, false);
        c1 = __builtin_amdgcn_wmma_f32_16x16x32_bf16(false, a, false, b1,
                                                     (short)0, c1, false, false);
    }

    // C layout: lanes 0-15 => N=lane, VGPR r => M=r ; lanes 16-31 => M=8+r
    const int mhi = (lane >> 4) * 8;
    #pragma unroll
    for (int r = 0; r < 8; ++r) {
        const int gm = m0 + wm * 16 + mhi + r;
        if (gm >= M) continue;
        #pragma unroll
        for (int sub = 0; sub < 2; ++sub) {
            const int gn = n0 + wn * 32 + sub * 16 + lm;
            if (gn >= N) continue;
            float v = sub ? c1[r] : c0[r];
            if (bias)  v += bias[gn];
            if (act)   v = gelu_exact(v);
            if (resid) v += resid[(size_t)gm * ldres + gn];
            if (C)   C[(size_t)gm * ldc + gn] = v;
            if (Cbf) Cbf[(size_t)gm * ldc + gn] = (__bf16)v;
        }
    }
}

// ---------------- fused attention: scores -> gate -> softmax over genes ------
__global__ __launch_bounds__(256) void attn_kernel(
    const float* __restrict__ Qs, const float* __restrict__ Qe, const float* __restrict__ Qt,
    const float* __restrict__ Ks, const float* __restrict__ Ke, const float* __restrict__ Kt,
    const float* __restrict__ gate_w, const float* __restrict__ gate_b,
    float* __restrict__ u_ws,        // [NTF, NG] scratch row per t
    __bf16* __restrict__ A_heads,    // [H, NTF, NG] bf16 (GEMM A operand)
    float* __restrict__ A_mean,      // [NTF, NG]     (d_out slice)
    float* __restrict__ u_mean,      // [NTF, NG]     (d_out slice)
    float* __restrict__ alpha_mean)  // [NTF, NG, 3]  (d_out slice)
{
    const int t = blockIdx.x;
    const int tid = threadIdx.x;
    __shared__ float q0[C_DK], q1[C_DK], q2[C_DK];
    __shared__ float red[256];
    float* urow = u_ws + (size_t)t * C_NG;
    const float rscale = 0.17677669529663687f;  // 1/sqrt(DK)
    const float invH = 1.0f / C_H;

    for (int h = 0; h < C_H; ++h) {
        if (tid < C_DK) {
            q0[tid] = Qs[(size_t)t * 128 + h * C_DK + tid];
            q1[tid] = Qe[(size_t)t * 128 + h * C_DK + tid];
            q2[tid] = Qt[(size_t)t * 128 + h * C_DK + tid];
        }
        __syncthreads();

        float gw[9], gb[3];
        #pragma unroll
        for (int i = 0; i < 9; ++i) gw[i] = gate_w[h * 9 + i];
        #pragma unroll
        for (int i = 0; i < 3; ++i) gb[i] = gate_b[h * 3 + i];

        const float4* qp0 = (const float4*)q0;
        const float4* qp1 = (const float4*)q1;
        const float4* qp2 = (const float4*)q2;

        float lmax = -3.0e38f;
        for (int g = tid; g < C_NG; g += 256) {
            const size_t ko = (size_t)g * 128 + h * C_DK;
            const float4* k0 = (const float4*)(Ks + ko);
            const float4* k1 = (const float4*)(Ke + ko);
            const float4* k2 = (const float4*)(Kt + ko);
            float s0 = 0.f, s1 = 0.f, s2 = 0.f;
            #pragma unroll
            for (int i = 0; i < 8; ++i) {
                float4 a = qp0[i], b = k0[i];
                s0 += a.x * b.x + a.y * b.y + a.z * b.z + a.w * b.w;
                a = qp1[i]; b = k1[i];
                s1 += a.x * b.x + a.y * b.y + a.z * b.z + a.w * b.w;
                a = qp2[i]; b = k2[i];
                s2 += a.x * b.x + a.y * b.y + a.z * b.z + a.w * b.w;
            }
            s0 *= rscale; s1 *= rscale; s2 *= rscale;

            float l0 = s0 * gw[0] + s1 * gw[3] + s2 * gw[6] + gb[0];
            float l1 = s0 * gw[1] + s1 * gw[4] + s2 * gw[7] + gb[1];
            float l2 = s0 * gw[2] + s1 * gw[5] + s2 * gw[8] + gb[2];
            float lm = fmaxf(l0, fmaxf(l1, l2));
            float e0 = expf(l0 - lm), e1 = expf(l1 - lm), e2 = expf(l2 - lm);
            float rs = 1.0f / (e0 + e1 + e2);
            float a0 = e0 * rs, a1 = e1 * rs, a2 = e2 * rs;
            float u = a0 * s0 + a1 * s1 + a2 * s2;

            urow[g] = u;
            lmax = fmaxf(lmax, u);

            const size_t o = (size_t)t * C_NG + g;
            if (h == 0) {
                u_mean[o] = u * invH;
                alpha_mean[o * 3 + 0] = a0 * invH;
                alpha_mean[o * 3 + 1] = a1 * invH;
                alpha_mean[o * 3 + 2] = a2 * invH;
            } else {
                u_mean[o] += u * invH;
                alpha_mean[o * 3 + 0] += a0 * invH;
                alpha_mean[o * 3 + 1] += a1 * invH;
                alpha_mean[o * 3 + 2] += a2 * invH;
            }
        }

        const float rmax = block_max256(lmax, red, tid);

        float lsum = 0.f;
        for (int g = tid; g < C_NG; g += 256) {   // same g-set per thread: no race
            float e = expf(urow[g] - rmax);
            urow[g] = e;
            lsum += e;
        }
        const float rinv = 1.0f / block_sum256(lsum, red, tid);

        __bf16* Ah = A_heads + ((size_t)h * C_NTF + t) * C_NG;
        for (int g = tid; g < C_NG; g += 256) {
            float a = urow[g] * rinv;
            Ah[g] = (__bf16)a;
            const size_t o = (size_t)t * C_NG + g;
            if (h == 0) A_mean[o] = a * invH;
            else        A_mean[o] += a * invH;
        }
        __syncthreads();
    }
}

// ---------------------------------------------------------------------------
extern "C" void kernel_launch(void* const* d_in, const int* in_sizes, int n_in,
                              void* d_out, int out_size, void* d_ws, size_t ws_size,
                              hipStream_t stream) {
    const float* H_TF   = (const float*)d_in[0];
    const float* H_G    = (const float*)d_in[1];
    const float* z_exp  = (const float*)d_in[2];
    const float* z_seq  = (const float*)d_in[3];
    const float* z_txt  = (const float*)d_in[4];
    const int*   tf_idx = (const int*)  d_in[5];
    const float* WQ_seq = (const float*)d_in[6];
    const float* WK_seq = (const float*)d_in[7];
    const float* WQ_exp = (const float*)d_in[8];
    const float* WK_exp = (const float*)d_in[9];
    const float* WQ_txt = (const float*)d_in[10];
    const float* WK_txt = (const float*)d_in[11];
    const float* gate_w = (const float*)d_in[12];
    const float* gate_b = (const float*)d_in[13];
    const float* WVG    = (const float*)d_in[14];
    const float* WVTF   = (const float*)d_in[15];
    const float* WOTF   = (const float*)d_in[16];
    const float* WOG    = (const float*)d_in[17];
    const float* ln_atf_g = (const float*)d_in[18];
    const float* ln_atf_b = (const float*)d_in[19];
    const float* ln_ag_g  = (const float*)d_in[20];
    const float* ln_ag_b  = (const float*)d_in[21];
    const float* ln_ftf_g = (const float*)d_in[22];
    const float* ln_ftf_b = (const float*)d_in[23];
    const float* ln_fg_g  = (const float*)d_in[24];
    const float* ln_fg_b  = (const float*)d_in[25];
    const float* ftf_W1 = (const float*)d_in[26];
    const float* ftf_b1 = (const float*)d_in[27];
    const float* ftf_W2 = (const float*)d_in[28];
    const float* ftf_b2 = (const float*)d_in[29];
    const float* fg_W1  = (const float*)d_in[30];
    const float* fg_b1  = (const float*)d_in[31];
    const float* fg_W2  = (const float*)d_in[32];
    const float* fg_b2  = (const float*)d_in[33];

    // ---- workspace (byte allocator, 256B aligned blocks) ----
    char* wsb = (char*)d_ws;
    size_t off = 0;
    auto walloc = [&](size_t bytes) {
        void* p = wsb + off;
        off += (bytes + 255) & ~(size_t)255;
        return p;
    };
    auto bfbuf = [&](size_t n) { return (__bf16*)walloc(n * sizeof(__bf16)); };
    auto f32buf = [&](size_t n) { return (float*)walloc(n * sizeof(float)); };

    // bf16 operands
    __bf16* zs_tf  = bfbuf((size_t)C_NTF * C_D);
    __bf16* ze_tf  = bfbuf((size_t)C_NTF * C_D);
    __bf16* zt_tf  = bfbuf((size_t)C_NTF * C_D);
    __bf16* zsq_b  = bfbuf((size_t)C_NG * C_D);
    __bf16* zex_b  = bfbuf((size_t)C_NG * C_D);
    __bf16* ztx_b  = bfbuf((size_t)C_NG * C_D);
    __bf16* HTFn_b = bfbuf((size_t)C_NTF * C_D);
    __bf16* HGn_b  = bfbuf((size_t)C_NG * C_D);
    __bf16* WQs_b  = bfbuf((size_t)128 * C_D);
    __bf16* WQe_b  = bfbuf((size_t)128 * C_D);
    __bf16* WQt_b  = bfbuf((size_t)128 * C_D);
    __bf16* WKs_b  = bfbuf((size_t)128 * C_D);
    __bf16* WKe_b  = bfbuf((size_t)128 * C_D);
    __bf16* WKt_b  = bfbuf((size_t)128 * C_D);
    __bf16* WVG_b  = bfbuf((size_t)C_D * C_D);
    __bf16* WVTF_b = bfbuf((size_t)C_D * C_D);
    __bf16* WOTF_b = bfbuf((size_t)C_D * C_D);
    __bf16* WOG_b  = bfbuf((size_t)C_D * C_D);
    __bf16* W1tf_b = bfbuf((size_t)C_DFF * C_D);
    __bf16* W2tf_b = bfbuf((size_t)C_D * C_DFF);
    __bf16* W1g_b  = bfbuf((size_t)C_DFF * C_D);
    __bf16* W2g_b  = bfbuf((size_t)C_D * C_DFF);
    __bf16* VG_b   = bfbuf((size_t)C_NG * C_D);
    __bf16* VTF_b  = bfbuf((size_t)C_NTF * C_D);
    __bf16* Ahd_b  = bfbuf((size_t)C_H * C_NTF * C_NG);
    __bf16* atf_b  = bfbuf((size_t)C_NTF * C_D);
    __bf16* ag_b   = bfbuf((size_t)C_NG * C_D);
    __bf16* xtf_b  = bfbuf((size_t)C_NTF * C_D);
    __bf16* xg_b   = bfbuf((size_t)C_NG * C_D);
    __bf16* gtf_b  = bfbuf((size_t)C_NTF * C_DFF);
    __bf16* gg_b   = bfbuf((size_t)C_NG * C_DFF);

    // fp32 intermediates
    float* Qseq = f32buf((size_t)C_NTF * 128);
    float* Qexp = f32buf((size_t)C_NTF * 128);
    float* Qtxt = f32buf((size_t)C_NTF * 128);
    float* Kseq = f32buf((size_t)C_NG * 128);
    float* Kexp = f32buf((size_t)C_NG * 128);
    float* Ktxt = f32buf((size_t)C_NG * 128);
    float* u_ws = f32buf((size_t)C_NTF * C_NG);
    float* HTF2 = f32buf((size_t)C_NTF * C_D);
    float* HG2  = f32buf((size_t)C_NG * C_D);

    // ---- output layout (return-order concat, fp32) ----
    float* out = (float*)d_out;
    float* oHTF3 = out;
    float* oHG3  = oHTF3 + (size_t)C_NTF * C_D;
    float* oAm   = oHG3  + (size_t)C_NG * C_D;
    float* oUm   = oAm   + (size_t)C_NTF * C_NG;
    float* oAl   = oUm   + (size_t)C_NTF * C_NG;

    auto cvt = [&](const float* x, __bf16* y, size_t n) {
        long n4 = (long)(n / 4);
        cvt_bf16_kernel<<<dim3((unsigned)((n4 + 255) / 256)), 256, 0, stream>>>(x, y, n4);
    };
    auto gemm = [&](int at, int bt, const __bf16* A, int lda, const __bf16* B, int ldb,
                    float* C, __bf16* Cbf, int ldc, const float* bias,
                    const float* res, int ldres, int M, int N, int K, int act) {
        dim3 grid((N + 63) / 64, (M + 63) / 64);
        if (!at && !bt)
            gemm_bf16_kernel<0,0><<<grid,256,0,stream>>>(A,lda,B,ldb,C,Cbf,ldc,bias,res,ldres,M,N,K,act);
        else if (!at && bt)
            gemm_bf16_kernel<0,1><<<grid,256,0,stream>>>(A,lda,B,ldb,C,Cbf,ldc,bias,res,ldres,M,N,K,act);
        else if (at && !bt)
            gemm_bf16_kernel<1,0><<<grid,256,0,stream>>>(A,lda,B,ldb,C,Cbf,ldc,bias,res,ldres,M,N,K,act);
        else
            gemm_bf16_kernel<1,1><<<grid,256,0,stream>>>(A,lda,B,ldb,C,Cbf,ldc,bias,res,ldres,M,N,K,act);
    };

    // 0) one-time bf16 conversions of raw inputs / weights
    cvt(z_seq, zsq_b, (size_t)C_NG * C_D);
    cvt(z_exp, zex_b, (size_t)C_NG * C_D);
    cvt(z_txt, ztx_b, (size_t)C_NG * C_D);
    cvt(WQ_seq, WQs_b, (size_t)128 * C_D);
    cvt(WQ_exp, WQe_b, (size_t)128 * C_D);
    cvt(WQ_txt, WQt_b, (size_t)128 * C_D);
    cvt(WK_seq, WKs_b, (size_t)128 * C_D);
    cvt(WK_exp, WKe_b, (size_t)128 * C_D);
    cvt(WK_txt, WKt_b, (size_t)128 * C_D);
    cvt(WVG,  WVG_b,  (size_t)C_D * C_D);
    cvt(WVTF, WVTF_b, (size_t)C_D * C_D);
    cvt(WOTF, WOTF_b, (size_t)C_D * C_D);
    cvt(WOG,  WOG_b,  (size_t)C_D * C_D);
    cvt(ftf_W1, W1tf_b, (size_t)C_DFF * C_D);
    cvt(ftf_W2, W2tf_b, (size_t)C_D * C_DFF);
    cvt(fg_W1,  W1g_b,  (size_t)C_DFF * C_D);
    cvt(fg_W2,  W2g_b,  (size_t)C_D * C_DFF);

    // 1) layernorms (bf16 outputs feed V projections)
    ln_kernel<<<C_NTF, 256, 0, stream>>>(H_TF, ln_atf_g, ln_atf_b, nullptr, HTFn_b);
    ln_kernel<<<C_NG,  256, 0, stream>>>(H_G,  ln_ag_g,  ln_ag_b,  nullptr, HGn_b);

    // 2) gather TF channel embeddings (bf16)
    gather3_kernel<<<C_NTF, 256, 0, stream>>>(z_seq, z_exp, z_txt, tf_idx,
                                              zs_tf, ze_tf, zt_tf);

    // 3) Q / K / V projections (y = x @ W.T, W stored [N,K] -> bt=0)
    gemm(0,0, zs_tf, C_D, WQs_b, C_D, Qseq, nullptr, 128, nullptr, nullptr, 0, C_NTF, 128, C_D, 0);
    gemm(0,0, ze_tf, C_D, WQe_b, C_D, Qexp, nullptr, 128, nullptr, nullptr, 0, C_NTF, 128, C_D, 0);
    gemm(0,0, zt_tf, C_D, WQt_b, C_D, Qtxt, nullptr, 128, nullptr, nullptr, 0, C_NTF, 128, C_D, 0);
    gemm(0,0, zsq_b, C_D, WKs_b, C_D, Kseq, nullptr, 128, nullptr, nullptr, 0, C_NG, 128, C_D, 0);
    gemm(0,0, zex_b, C_D, WKe_b, C_D, Kexp, nullptr, 128, nullptr, nullptr, 0, C_NG, 128, C_D, 0);
    gemm(0,0, ztx_b, C_D, WKt_b, C_D, Ktxt, nullptr, 128, nullptr, nullptr, 0, C_NG, 128, C_D, 0);
    gemm(0,0, HGn_b,  C_D, WVG_b,  C_D, nullptr, VG_b,  C_D, nullptr, nullptr, 0, C_NG,  C_D, C_D, 0);
    gemm(0,0, HTFn_b, C_D, WVTF_b, C_D, nullptr, VTF_b, C_D, nullptr, nullptr, 0, C_NTF, C_D, C_D, 0);

    // 4) fused scores + gate + gene-softmax + mean outputs
    attn_kernel<<<C_NTF, 256, 0, stream>>>(Qseq, Qexp, Qtxt, Kseq, Kexp, Ktxt,
                                           gate_w, gate_b, u_ws, Ahd_b,
                                           oAm, oUm, oAl);

    // 5) attention-value contractions per head
    for (int h = 0; h < C_H; ++h) {
        const __bf16* Ah = Ahd_b + (size_t)h * C_NTF * C_NG;
        // out_tf[t, h*64+v] = sum_g A_h[t,g] * V_G[g, h*64+v]
        gemm(0,1, Ah, C_NG, VG_b + h * 64, C_D,
             nullptr, atf_b + h * 64, C_D, nullptr, nullptr, 0, C_NTF, 64, C_NG, 0);
        // out_g[g, h*64+v] = sum_t A_h[t,g] * V_TF[t, h*64+v]
        gemm(1,1, Ah, C_NG, VTF_b + h * 64, C_D,
             nullptr, ag_b + h * 64, C_D, nullptr, nullptr, 0, C_NG, 64, C_NTF, 0);
    }

    // 6) output projections + residual
    gemm(0,0, atf_b, C_D, WOTF_b, C_D, HTF2, nullptr, C_D, nullptr, H_TF, C_D, C_NTF, C_D, C_D, 0);
    gemm(0,0, ag_b,  C_D, WOG_b,  C_D, HG2,  nullptr, C_D, nullptr, H_G,  C_D, C_NG,  C_D, C_D, 0);

    // 7) FFN blocks (LN -> W1+b1 -> exact GELU -> W2+b2 -> +residual)
    ln_kernel<<<C_NTF, 256, 0, stream>>>(HTF2, ln_ftf_g, ln_ftf_b, nullptr, xtf_b);
    ln_kernel<<<C_NG,  256, 0, stream>>>(HG2,  ln_fg_g,  ln_fg_b,  nullptr, xg_b);
    gemm(0,0, xtf_b, C_D, W1tf_b, C_D, nullptr, gtf_b, C_DFF, ftf_b1, nullptr, 0,
         C_NTF, C_DFF, C_D, 1);
    gemm(0,0, gtf_b, C_DFF, W2tf_b, C_DFF, oHTF3, nullptr, C_D, ftf_b2, HTF2, C_D,
         C_NTF, C_D, C_DFF, 0);
    gemm(0,0, xg_b, C_D, W1g_b, C_D, nullptr, gg_b, C_DFF, fg_b1, nullptr, 0,
         C_NG, C_DFF, C_D, 1);
    gemm(0,0, gg_b, C_DFF, W2g_b, C_DFF, oHG3, nullptr, C_D, fg_b2, HG2, C_D,
         C_NG, C_D, C_DFF, 0);
}
